// BlockPGA_90013924590445
// MI455X (gfx1250) — matrix operand order repaired and checked by reading, not compiled, not run on hardware
//
#include <hip/hip_runtime.h>
#include <hip/hip_bf16.h>

typedef __attribute__((ext_vector_type(16))) _Float16 v16h;
typedef __attribute__((ext_vector_type(8)))  float    v8f;

#define HW   90000
#define EMB  64
#define DH   32
#define CROP 300
#define CIN  256

__device__ __forceinline__ v8f vzero8(){ v8f z = {}; return z; }

__device__ __forceinline__ v8f wmma16(v16h a, v16h b, v8f c){
  // v_wmma_f32_16x16x32_f16 : D = A(16x32) * B(32x16) + C(16x16 f32)
  return __builtin_amdgcn_wmma_f32_16x16x32_f16(false, a, false, b, (short)0, c, false, false);
}

// ---- fragment loaders (wave32 CDNA5 WMMA layouts) ----
// A-frag 16x32 f16: lane L holds row, k = (i<8?0:16) + (L/16)*8 + i%8
__device__ __forceinline__ v16h a_from_f32(const float* A, int row, int stride, int koff){
  const int lh = (threadIdx.x >> 4) & 1;
  const float* p = A + (size_t)row * stride + koff + lh * 8;
  v16h a;
#pragma unroll
  for (int i = 0; i < 8; i++) a[i]     = (_Float16)p[i];
#pragma unroll
  for (int i = 0; i < 8; i++) a[8 + i] = (_Float16)p[16 + i];
  return a;
}
__device__ __forceinline__ v16h a_from_h(const _Float16* A, int row, int stride, int koff){
  const int lh = (threadIdx.x >> 4) & 1;
  const _Float16* p = A + (size_t)row * stride + koff + lh * 8;
  v16h a;
#pragma unroll
  for (int i = 0; i < 8; i++) a[i]     = p[i];
#pragma unroll
  for (int i = 0; i < 8; i++) a[8 + i] = p[16 + i];
  return a;
}
// B-frag 32x16 f16 from storage S[n][k] (row-major, stride halves): B[k][n]=S[n][k]
// lane L: n fixed, k = (i<8?0:16) + (L/16)*8 + i%8
__device__ __forceinline__ v16h b_from_h(const _Float16* S, int n, int stride, int koff){
  const int lh = (threadIdx.x >> 4) & 1;
  const _Float16* p = S + (size_t)n * stride + koff + lh * 8;
  v16h b;
#pragma unroll
  for (int i = 0; i < 8; i++) b[i] = p[i];
#pragma unroll
  for (int i = 0; i < 8; i++) b[8 + i] = p[16 + i];
  return b;
}
// B-frag from f32 matrix stored [k][n] row-major with row stride rs: B[k][n] = M[(koff+k)*rs + n]
__device__ __forceinline__ v16h b_from_f32_kn(const float* M, int n, int rs, int koff){
  const int lh = (threadIdx.x >> 4) & 1;
  const float* p = M + (size_t)(koff + lh * 8) * rs + n;
  v16h b;
#pragma unroll
  for (int i = 0; i < 8; i++) b[i] = (_Float16)p[(size_t)i * rs];
#pragma unroll
  for (int i = 0; i < 8; i++) b[8 + i] = (_Float16)p[(size_t)(16 + i) * rs];
  return b;
}

__device__ __forceinline__ float redmax16(float v){
  v = fmaxf(v, __shfl_xor(v, 1, 32));
  v = fmaxf(v, __shfl_xor(v, 2, 32));
  v = fmaxf(v, __shfl_xor(v, 4, 32));
  v = fmaxf(v, __shfl_xor(v, 8, 32));
  return v;
}
__device__ __forceinline__ float redsum16(float v){
  v += __shfl_xor(v, 1, 32);
  v += __shfl_xor(v, 2, 32);
  v += __shfl_xor(v, 4, 32);
  v += __shfl_xor(v, 8, 32);
  return v;
}

// ============ K1: conv1 (x1pre = W1 @ x) + BN1 channel sums ============
__global__ void __launch_bounds__(256) k_conv1(const float* __restrict__ x,
                                               const float* __restrict__ w1,
                                               float* __restrict__ x1,
                                               float* __restrict__ gsum,
                                               float* __restrict__ gsq){
  __shared__ float ssum[64], ssq[64];
  const int tid = threadIdx.x;
  if (tid < 64){ ssum[tid] = 0.f; ssq[tid] = 0.f; }
  __syncthreads();
  const int wave = tid >> 5;
  const int lr = tid & 15, lh = (tid >> 4) & 1;
  const int mt = wave & 3;                       // 4 M-tiles cover 64 channels
  const int ntile = blockIdx.x * 2 + (wave >> 2);
  if (ntile < HW / 16){
    const int t = ntile * 16 + lr;
    v8f acc = vzero8();
    for (int kt = 0; kt < CIN / 32; kt++){
      v16h a = a_from_f32(w1, mt * 16 + lr, CIN, kt * 32);
      v16h b = b_from_f32_kn(x, t, HW, kt * 32);
      acc = wmma16(a, b, acc);
    }
#pragma unroll
    for (int r = 0; r < 8; r++){
      int ch = mt * 16 + lh * 8 + r;
      float v = acc[r];
      x1[(size_t)ch * HW + t] = v;
      atomicAdd(&ssum[ch], v);
      atomicAdd(&ssq[ch], v * v);
    }
  }
  __syncthreads();
  if (tid < 64){ atomicAdd(&gsum[tid], ssum[tid]); atomicAdd(&gsq[tid], ssq[tid]); }
}

// ============ K2: BN finalize (64 threads) ============
__global__ void k_bnstat(const float* __restrict__ sum, const float* __restrict__ sq,
                         const float* __restrict__ g, const float* __restrict__ b,
                         float* __restrict__ scale, float* __restrict__ shift){
  int i = threadIdx.x;
  float mean = sum[i] * (1.0f / HW);
  float var  = sq[i]  * (1.0f / HW) - mean * mean;
  float sc = g[i] * rsqrtf(var + 1e-5f);
  scale[i] = sc;
  shift[i] = b[i] - mean * sc;
}

// ============ K3: apply BN + relu ============
__global__ void k_bnapply(const float* __restrict__ in, float* __restrict__ out,
                          const float* __restrict__ scale, const float* __restrict__ shift){
  int i = blockIdx.x * blockDim.x + threadIdx.x;
  if (i < EMB * HW){
    int c = i / HW;
    out[i] = fmaxf(in[i] * scale[c] + shift[c], 0.0f);
  }
}

// ============ K4: prop attention (600 blocks: head x row) ============
__global__ void __launch_bounds__(256) k_attn(const float* __restrict__ x1,
                                              const float* __restrict__ wq,
                                              const float* __restrict__ wkv,
                                              const int* __restrict__ obj,
                                              const int* __restrict__ bg,
                                              const int* __restrict__ rinds,
                                              float* __restrict__ img){
  __shared__ _Float16 Km[320 * 32];      // K  [token][d]
  __shared__ _Float16 Vt[32 * 320];      // V^T[d][token]
  __shared__ _Float16 wql[32 * 32], wkl[32 * 32], wvl[32 * 32]; // [n][k]
  __shared__ int      posv[320];
  __shared__ _Float16 pbuf[8][16 * 32];  // per-wave D->A bounce tile

  const int tid = threadIdx.x;
  const int h = blockIdx.x / CROP, a = blockIdx.x % CROP;

  // ---- phase A: weights to LDS (f16), pos table, zero pads ----
  for (int i = tid; i < 32 * 32; i += 256){
    wql[i] = (_Float16)wq[i];
    wkl[i] = (_Float16)wkv[i];
    wvl[i] = (_Float16)wkv[32 * 32 + i];
  }
  for (int c = tid; c < 320; c += 256){
    int p = 0;
    if (c < CROP){
      int ri = rinds[((h * CROP) + a) * CROP + c];
      p = (a < CROP / 2) ? obj[ri] : bg[ri];
    }
    posv[c] = p;
  }
  for (int i = tid; i < 16 * 32; i += 256){   // zero tokens 304..319
    int tok = 304 + (i >> 5), d = i & 31;
    Km[tok * 32 + d]  = (_Float16)0.f;
    Vt[d * 320 + tok] = (_Float16)0.f;
  }
  __syncthreads();

  const int wave = tid >> 5;
  const int lr = tid & 15, lh = (tid >> 4) & 1;
  _Float16* pb = &pbuf[wave][0];

  // ---- phase B: K = T @ Wk^T, V = T @ Wv^T (19 M-tiles x {K,V}) ----
  for (int task = wave; task < 38; task += 8){
    int mt  = task % 19;
    int isv = task / 19;
    // gather A-frag of T directly from global (x1 lives in 192MB L2)
    v16h afr;
    {
      int tok = mt * 16 + lr;
      const float* xb = x1 + (size_t)(h * DH) * HW + posv[tok];
#pragma unroll
      for (int i = 0; i < 8; i++) afr[i]     = (_Float16)xb[(size_t)(lh * 8 + i) * HW];
#pragma unroll
      for (int i = 0; i < 8; i++) afr[8 + i] = (_Float16)xb[(size_t)(16 + lh * 8 + i) * HW];
    }
    const _Float16* wmat = isv ? wvl : wkl;
    for (int nt = 0; nt < 2; nt++){
      v16h bfr = b_from_h(wmat, nt * 16 + lr, 32, 0);
      v8f d = wmma16(afr, bfr, vzero8());
#pragma unroll
      for (int r = 0; r < 8; r++){
        int tok = mt * 16 + lh * 8 + r;
        int dd  = nt * 16 + lr;
        if (isv) Vt[dd * 320 + tok] = (_Float16)d[r];
        else     Km[tok * 32 + dd]  = (_Float16)d[r];
      }
    }
  }
  __syncthreads();

  // ---- phase C: per-wave query tiles, flash-style softmax ----
  const float dscale = 0.17677669529663687f; // 1/sqrt(32)
  for (int mt = wave; mt < 19; mt += 8){
    // Q tile: T @ Wq^T, bounced via LDS into A-frag
    v16h qfr;
    {
      v16h afr;
      int tok = mt * 16 + lr;
      const float* xb = x1 + (size_t)(h * DH) * HW + posv[tok];
#pragma unroll
      for (int i = 0; i < 8; i++) afr[i]     = (_Float16)xb[(size_t)(lh * 8 + i) * HW];
#pragma unroll
      for (int i = 0; i < 8; i++) afr[8 + i] = (_Float16)xb[(size_t)(16 + lh * 8 + i) * HW];
      for (int nt = 0; nt < 2; nt++){
        v16h bfr = b_from_h(wql, nt * 16 + lr, 32, 0);
        v8f d = wmma16(afr, bfr, vzero8());
#pragma unroll
        for (int r = 0; r < 8; r++) pb[(lh * 8 + r) * 32 + nt * 16 + lr] = (_Float16)d[r];
      }
      asm volatile("s_wait_dscnt 0" ::: "memory");
      qfr = a_from_h(pb, lr, 32, 0);
    }

    v8f acc0 = vzero8(), acc1 = vzero8();
    float mrow[8], lrow[8];
#pragma unroll
    for (int r = 0; r < 8; r++){ mrow[r] = -1e30f; lrow[r] = 0.f; }

    for (int s = 0; s < 10; s++){               // 10 strips of 32 keys (padded to 320)
      int tA = s * 32;
      v16h kb0 = b_from_h(Km, tA + lr, 32, 0);
      v16h kb1 = b_from_h(Km, tA + 16 + lr, 32, 0);
      v8f s0 = wmma16(qfr, kb0, vzero8());
      v8f s1 = wmma16(qfr, kb1, vzero8());
      const bool ok0 = (tA + lr) < CROP, ok1 = (tA + 16 + lr) < CROP;
      float p0[8], p1[8], fac[8];
#pragma unroll
      for (int r = 0; r < 8; r++){
        float a0 = ok0 ? s0[r] * dscale : -1e30f;
        float a1 = ok1 ? s1[r] * dscale : -1e30f;
        float sm = redmax16(fmaxf(a0, a1));
        float mn = fmaxf(mrow[r], sm);
        fac[r] = __expf(mrow[r] - mn);
        p0[r]  = __expf(a0 - mn);
        p1[r]  = __expf(a1 - mn);
        float rs = redsum16(p0[r] + p1[r]);
        lrow[r] = lrow[r] * fac[r] + rs;
        mrow[r] = mn;
      }
#pragma unroll
      for (int r = 0; r < 8; r++){ acc0[r] *= fac[r]; acc1[r] *= fac[r]; }
      // p -> LDS -> A-frag
#pragma unroll
      for (int r = 0; r < 8; r++){
        pb[(lh * 8 + r) * 32 + lr]      = (_Float16)p0[r];
        pb[(lh * 8 + r) * 32 + 16 + lr] = (_Float16)p1[r];
      }
      asm volatile("s_wait_dscnt 0" ::: "memory");
      v16h pa  = a_from_h(pb, lr, 32, 0);
      v16h vb0 = b_from_h(Vt, lr,      320, tA);
      v16h vb1 = b_from_h(Vt, 16 + lr, 320, tA);
      acc0 = wmma16(pa, vb0, acc0);
      acc1 = wmma16(pa, vb1, acc1);
    }

    // normalize + scrambled scatter: value out[tok][dd] -> m=tok*32+dd; d=m/300,c=m%300
#pragma unroll
    for (int r = 0; r < 8; r++){
      int tok = mt * 16 + lh * 8 + r;
      if (tok < CROP){
        float inv = 1.0f / lrow[r];
        int m0 = tok * 32 + lr;
        int d0 = m0 / CROP, c0 = m0 % CROP;
        img[(size_t)(h * DH + d0) * HW + posv[c0]] = acc0[r] * inv;
        int m1 = tok * 32 + 16 + lr;
        int d1 = m1 / CROP, c1 = m1 % CROP;
        img[(size_t)(h * DH + d1) * HW + posv[c1]] = acc1[r] * inv;
      }
    }
  }
}

// ============ K5: pre2 = W2a @ relu(Wo@img + bo) + W2b @ x1, + BN2 sums ============
__global__ void __launch_bounds__(256) k_out(const float* __restrict__ img,
                                             const float* __restrict__ x1,
                                             const float* __restrict__ wo,
                                             const float* __restrict__ bo,
                                             const float* __restrict__ w2,
                                             float* __restrict__ pre2,
                                             float* __restrict__ gsum,
                                             float* __restrict__ gsq){
  __shared__ _Float16 wol[64 * 64];
  __shared__ _Float16 w2l[64 * 128];
  __shared__ _Float16 ub[8][16 * 64];   // per-wave u tile, [pixel][ch]
  __shared__ float ssum[64], ssq[64];
  const int tid = threadIdx.x;
  for (int i = tid; i < 64 * 64;  i += 256) wol[i] = (_Float16)wo[i];
  for (int i = tid; i < 64 * 128; i += 256) w2l[i] = (_Float16)w2[i];
  if (tid < 64){ ssum[tid] = 0.f; ssq[tid] = 0.f; }
  __syncthreads();

  const int wave = tid >> 5, lr = tid & 15, lh = (tid >> 4) & 1;
  const int t0 = blockIdx.x * 128 + wave * 16;
  if (t0 < HW){
    const int t = min(t0 + lr, HW - 1);
    const bool valid = (t0 + lr) < HW;
    _Float16* ubw = &ub[wave][0];
    v16h bi0 = b_from_f32_kn(img, t, HW, 0);
    v16h bi1 = b_from_f32_kn(img, t, HW, 32);
    // stage 1: u = relu(Wo @ img + bo)
    for (int mt = 0; mt < 4; mt++){
      v8f acc = wmma16(a_from_h(wol, mt * 16 + lr, 64, 0),  bi0, vzero8());
      acc     = wmma16(a_from_h(wol, mt * 16 + lr, 64, 32), bi1, acc);
#pragma unroll
      for (int r = 0; r < 8; r++){
        int ch = mt * 16 + lh * 8 + r;
        float u = fmaxf(acc[r] + bo[ch], 0.f);
        ubw[lr * 64 + ch] = (_Float16)u;
      }
    }
    asm volatile("s_wait_dscnt 0" ::: "memory");
    // stage 2: W2 @ concat(u, x1)
    v16h b20 = b_from_h(ubw, lr, 64, 0);
    v16h b21 = b_from_h(ubw, lr, 64, 32);
    v16h b22 = b_from_f32_kn(x1, t, HW, 0);
    v16h b23 = b_from_f32_kn(x1, t, HW, 32);
    for (int mt = 0; mt < 4; mt++){
      v8f acc = vzero8();
      acc = wmma16(a_from_h(w2l, mt * 16 + lr, 128, 0),  b20, acc);
      acc = wmma16(a_from_h(w2l, mt * 16 + lr, 128, 32), b21, acc);
      acc = wmma16(a_from_h(w2l, mt * 16 + lr, 128, 64), b22, acc);
      acc = wmma16(a_from_h(w2l, mt * 16 + lr, 128, 96), b23, acc);
#pragma unroll
      for (int r = 0; r < 8; r++){
        int ch = mt * 16 + lh * 8 + r;
        float v = acc[r];
        if (valid){
          pre2[(size_t)ch * HW + t0 + lr] = v;
          atomicAdd(&ssum[ch], v);
          atomicAdd(&ssq[ch], v * v);
        }
      }
    }
  }
  __syncthreads();
  if (tid < 64){ atomicAdd(&gsum[tid], ssum[tid]); atomicAdd(&gsq[tid], ssq[tid]); }
}

extern "C" void kernel_launch(void* const* d_in, const int* in_sizes, int n_in,
                              void* d_out, int out_size, void* d_ws, size_t ws_size,
                              hipStream_t stream){
  const float* x      = (const float*)d_in[0];
  const float* w1     = (const float*)d_in[1];
  const float* bn1g   = (const float*)d_in[2];
  const float* bn1b   = (const float*)d_in[3];
  const float* wq     = (const float*)d_in[4];
  const float* wkv    = (const float*)d_in[5];
  const float* wo     = (const float*)d_in[6];
  const float* bo     = (const float*)d_in[7];
  const float* w2     = (const float*)d_in[8];
  const float* bn2g   = (const float*)d_in[9];
  const float* bn2b   = (const float*)d_in[10];
  const int*   obj    = (const int*)d_in[11];
  const int*   bg     = (const int*)d_in[12];
  const int*   rinds  = (const int*)d_in[13];
  float* out = (float*)d_out;

  float* ws    = (float*)d_ws;
  float* x1    = ws;                     // 64*90000
  float* img   = ws + (size_t)EMB * HW;  // 64*90000
  float* pre2  = img + (size_t)EMB * HW; // 64*90000
  float* stats = pre2 + (size_t)EMB * HW;
  float* sum1 = stats, *sq1 = stats + 64, *sum2 = stats + 128, *sq2 = stats + 192;
  float* sc1 = stats + 256, *sh1 = stats + 320, *sc2 = stats + 384, *sh2 = stats + 448;

  hipMemsetAsync(stats, 0, 256 * sizeof(float), stream);

  k_conv1<<<(HW / 16 + 1) / 2, 256, 0, stream>>>(x, w1, x1, sum1, sq1);
  k_bnstat<<<1, 64, 0, stream>>>(sum1, sq1, bn1g, bn1b, sc1, sh1);
  k_bnapply<<<(EMB * HW + 255) / 256, 256, 0, stream>>>(x1, x1, sc1, sh1);
  k_attn<<<2 * CROP, 256, 0, stream>>>(x1, wq, wkv, obj, bg, rinds, img);
  k_out<<<(HW + 127) / 128, 256, 0, stream>>>(img, x1, wo, bo, w2, pre2, sum2, sq2);
  k_bnstat<<<1, 64, 0, stream>>>(sum2, sq2, bn2g, bn2b, sc2, sh2);
  k_bnapply<<<(EMB * HW + 255) / 256, 256, 0, stream>>>(pre2, out, sc2, sh2);
}
